// GPTModel_89567247991654
// MI455X (gfx1250) — compile-verified, matrix operand order checked by reading, not run on hardware
//
#include <hip/hip_runtime.h>
#include <hip/hip_bf16.h>

// ---------------------------------------------------------------------------
// GPT forward for MI455X (gfx1250).
// GEMMs: v_wmma_f32_16x16x32_f16 fed from LDS, with double-buffered
// global->LDS async copies (global_load_async_to_lds_b128, ASYNCcnt).
// Block tile 64x64, 2 output tiles per wave, K-stage 64 (4 WMMA / barrier).
// B=1, S=1024, D=1024, H=16, HD=64, L=4, FF=4096, V=32000.
// ---------------------------------------------------------------------------

#define S_  1024
#define D_  1024
#define H_  16
#define HD_ 64
#define L_  4
#define FF_ 4096
#define V_  32000
#define EPS_ 1e-5f

typedef __attribute__((ext_vector_type(16))) _Float16 v16h;
typedef __attribute__((ext_vector_type(8)))  _Float16 v8h;
typedef __attribute__((ext_vector_type(8)))  float    v8f;

union Frag16 { v16h v; v8h h2[2]; };

// Low 32 bits of a generic shared-space pointer == LDS byte offset (ISA 10.2).
__device__ __forceinline__ unsigned lds_off(const void* p) {
    return (unsigned)(size_t)p;
}

#define PAD_ 72   // 64 data halves + 8 pad -> 144B row stride, conflict-free frags

// ---------------------------------------------------------------------------
// Generic f16 WMMA GEMM:  C[z] = act( scale * A[z] @ Bp[z]^T + bias + residual )
//   A  : M x K row-major f16 (row stride lda, batch stride sA elements)
//   Bp : N x K row-major f16 (weights pre-transposed; row stride ldb)
//   Cf : optional f32 output, Ch: optional f16 output (row stride ldc)
// Block = 256 threads = 8 wave32. Block tile 64(M) x 64(N); each wave owns
// two 16x16 tiles (shared B fragment). K-stage 64 halves, double buffered in
// LDS via async global->LDS copies: every thread issues exactly 4 b128 async
// ops per stage, so steady-state dependency is s_wait_asynccnt 4.
// M,N multiples of 64; K multiple of 64 for every launch.
// ---------------------------------------------------------------------------
__global__ __launch_bounds__(256) void gemm_wmma_f16(
    const _Float16* __restrict__ A, const _Float16* __restrict__ B,
    const float* __restrict__ bias, const float* __restrict__ residual,
    float* __restrict__ Cf, _Float16* __restrict__ Ch,
    int M, int N, int K, int lda, int ldb, int ldc,
    long sA, long sB, long sC, float scale, int relu)
{
    __shared__ _Float16 As[2][64][PAD_];   // 18 KB
    __shared__ _Float16 Bs[2][64][PAD_];   // 18 KB

    const int z    = blockIdx.z;
    const int t    = threadIdx.x;
    const int wid  = t >> 5;               // wave id 0..7 (wave32)
    const int lane = t & 31;
    const int mblk = blockIdx.y * 64;
    const int nblk = blockIdx.x * 64;

    const int r    = lane & 15;
    const int khi  = lane >> 4;            // lane-half -> K sub-offset (+8)
    const int mrow = (wid >> 2) * 32;      // 0 / 32
    const int ncol = (wid & 3) * 16;       // 0,16,32,48
    const int ar0  = mrow + r;             // A-frag row, tile 0
    const int ar1  = mrow + 16 + r;        // A-frag row, tile 1
    const int br   = ncol + r;             // B-frag row

    // Async-copy assignment: panels are 64 rows x 128B; 512 b128 chunks each.
    // Thread t owns chunks {t, t+256} of A and of B -> 4 asyncs/stage/thread.
    const _Float16* Abase = A + z * sA;
    const _Float16* Bbase = B + z * sB;

    auto issue_stage = [&](int buf, int k0) {
#pragma unroll
        for (int p = 0; p < 2; ++p) {
            const int c    = t + p * 256;
            const int row  = c >> 3;            // 0..63
            const int offh = (c & 7) * 8;       // half offset within row: 0..56
            unsigned lA = lds_off(&As[buf][row][offh]);
            unsigned lB = lds_off(&Bs[buf][row][offh]);
            const _Float16* gA = Abase + (long)(mblk + row) * lda + k0 + offh;
            const _Float16* gB = Bbase + (long)(nblk + row) * ldb + k0 + offh;
            asm volatile("global_load_async_to_lds_b128 %0, %1, off"
                         :: "v"(lA), "v"((unsigned long long)(size_t)gA) : "memory");
            asm volatile("global_load_async_to_lds_b128 %0, %1, off"
                         :: "v"(lB), "v"((unsigned long long)(size_t)gB) : "memory");
        }
    };

    v8f acc0 = {}, acc1 = {};

    auto compute_stage = [&](int buf) {
#pragma unroll
        for (int ks = 0; ks < 64; ks += 32) {
            Frag16 a0, a1, b;
            const v8h* A0 = (const v8h*)&As[buf][ar0][ks];
            const v8h* A1 = (const v8h*)&As[buf][ar1][ks];
            const v8h* Bf = (const v8h*)&Bs[buf][br][ks];
            a0.h2[0] = A0[khi];  a0.h2[1] = A0[khi + 2];
            a1.h2[0] = A1[khi];  a1.h2[1] = A1[khi + 2];
            b.h2[0]  = Bf[khi];  b.h2[1]  = Bf[khi + 2];
            acc0 = __builtin_amdgcn_wmma_f32_16x16x32_f16(
                false, a0.v, false, b.v, (short)0, acc0, false, false);
            acc1 = __builtin_amdgcn_wmma_f32_16x16x32_f16(
                false, a1.v, false, b.v, (short)0, acc1, false, false);
        }
    };

    const int nStage = K >> 6;
    int buf = 0;
    issue_stage(0, 0);
    // steady state: prefetch stage it+1, wait for stage it, compute
    for (int it = 0; it < nStage - 1; ++it) {
        issue_stage(buf ^ 1, (it + 1) << 6);
        asm volatile("s_wait_asynccnt 0x4" ::: "memory");  // stage `it` complete
        __syncthreads();
        compute_stage(buf);
        __syncthreads();        // reads done before buf is refilled at it+2
        buf ^= 1;
    }
    // epilogue: last stage
    asm volatile("s_wait_asynccnt 0x0" ::: "memory");
    __syncthreads();
    compute_stage(buf);

    // C/D layout: VGPR i -> row base+i (lanes 0-15) / base+8+i (lanes 16-31)
    const int col  = nblk + ncol + r;
    const int row0 = mblk + mrow + khi * 8;
    const float bv = bias ? bias[col] : 0.f;
#pragma unroll
    for (int i = 0; i < 8; ++i) {
        {   // tile 0
            const long idx = z * sC + (long)(row0 + i) * ldc + col;
            float v = acc0[i] * scale + bv;
            if (residual) v += residual[idx];
            if (relu) v = fmaxf(v, 0.f);
            if (Cf) Cf[idx] = v;
            if (Ch) Ch[idx] = (_Float16)v;
        }
        {   // tile 1 (+16 rows)
            const long idx = z * sC + (long)(row0 + 16 + i) * ldc + col;
            float v = acc1[i] * scale + bv;
            if (residual) v += residual[idx];
            if (relu) v = fmaxf(v, 0.f);
            if (Cf) Cf[idx] = v;
            if (Ch) Ch[idx] = (_Float16)v;
        }
    }
}

// ---------------------------------------------------------------------------
// Weight pack: out[z][n*M + m] = (f16) in[z][m*N + n]   (LDS-tiled transpose)
// ---------------------------------------------------------------------------
__global__ __launch_bounds__(256) void transpose_pack_f32(
    const float* __restrict__ in, _Float16* __restrict__ out,
    int M, int N, long sIn, long sOut)
{
    __shared__ _Float16 tile[32][33];
    const float* inp  = in  + blockIdx.z * sIn;
    _Float16*    outp = out + blockIdx.z * sOut;
    const int bm = blockIdx.y * 32, bn = blockIdx.x * 32;
    const int tx = threadIdx.x;
    for (int i = threadIdx.y; i < 32; i += blockDim.y)
        tile[i][tx] = (_Float16)inp[(long)(bm + i) * N + (bn + tx)];
    __syncthreads();
    for (int i = threadIdx.y; i < 32; i += blockDim.y)
        outp[(long)(bn + i) * M + (bm + tx)] = tile[tx][i];
}

// f16 -> f16 transpose with explicit leading dims (per-head V transpose)
__global__ __launch_bounds__(256) void transpose_f16(
    const _Float16* __restrict__ in, _Float16* __restrict__ out,
    int ldi, int ldo, long sIn, long sOut)
{
    __shared__ _Float16 tile[32][33];
    const _Float16* inp  = in  + blockIdx.z * sIn;
    _Float16*       outp = out + blockIdx.z * sOut;
    const int bm = blockIdx.y * 32, bn = blockIdx.x * 32;
    const int tx = threadIdx.x;
    for (int i = threadIdx.y; i < 32; i += blockDim.y)
        tile[i][tx] = inp[(long)(bm + i) * ldi + (bn + tx)];
    __syncthreads();
    for (int i = threadIdx.y; i < 32; i += blockDim.y)
        outp[(long)(bn + i) * ldo + (bm + tx)] = tile[tx][i];
}

// ---------------------------------------------------------------------------
// h[s,d] = emb[x[s], d] + sinusoidal_pe(s, d)
// ---------------------------------------------------------------------------
__global__ __launch_bounds__(256) void embed_kernel(
    const int* __restrict__ x, const float* __restrict__ emb,
    float* __restrict__ h)
{
    const int idx = blockIdx.x * blockDim.x + threadIdx.x;
    if (idx >= S_ * D_) return;
    const int s = idx / D_, d = idx % D_;
    const float inv = -logf(10000.f) / (float)D_;
    const float ang = (float)s * __expf((float)(d & ~1) * inv);
    const float pe  = (d & 1) ? __cosf(ang) : __sinf(ang);
    h[idx] = emb[(long)x[s] * D_ + d] + pe;
}

// ---------------------------------------------------------------------------
// LayerNorm (block per row); writes f16 for the following WMMA GEMMs
// ---------------------------------------------------------------------------
__global__ __launch_bounds__(256) void layernorm_f16(
    const float* __restrict__ x, const float* __restrict__ w,
    const float* __restrict__ b, _Float16* __restrict__ out, int D)
{
    __shared__ float s1[256], s2[256];
    const int row = blockIdx.x, tid = threadIdx.x;
    const float* xr = x + (long)row * D;
    float sum = 0.f, sq = 0.f;
    for (int d = tid; d < D; d += 256) { float v = xr[d]; sum += v; sq += v * v; }
    s1[tid] = sum; s2[tid] = sq; __syncthreads();
    for (int off = 128; off; off >>= 1) {
        if (tid < off) { s1[tid] += s1[tid + off]; s2[tid] += s2[tid + off]; }
        __syncthreads();
    }
    const float mean = s1[0] / D;
    const float var  = s2[0] / D - mean * mean;
    const float rstd = rsqrtf(var + EPS_);
    for (int d = tid; d < D; d += 256)
        out[(long)row * D + d] = (_Float16)((xr[d] - mean) * rstd * w[d] + b[d]);
}

// ---------------------------------------------------------------------------
// Causal softmax over scores[head][q][*] -> attn f16 (masked cols -> 0)
// ---------------------------------------------------------------------------
__global__ __launch_bounds__(256) void softmax_causal(
    const float* __restrict__ sc, _Float16* __restrict__ at)
{
    __shared__ float red[256];
    const int q = blockIdx.x, hz = blockIdx.y, tid = threadIdx.x;
    const float* row  = sc + ((long)hz * S_ + q) * S_;
    _Float16*    orow = at + ((long)hz * S_ + q) * S_;

    float m = -3.0e38f;
    for (int j = tid; j <= q; j += 256) m = fmaxf(m, row[j]);
    red[tid] = m; __syncthreads();
    for (int off = 128; off; off >>= 1) {
        if (tid < off) red[tid] = fmaxf(red[tid], red[tid + off]);
        __syncthreads();
    }
    m = red[0]; __syncthreads();

    float sum = 0.f;
    for (int j = tid; j <= q; j += 256) sum += __expf(row[j] - m);
    red[tid] = sum; __syncthreads();
    for (int off = 128; off; off >>= 1) {
        if (tid < off) red[tid] += red[tid + off];
        __syncthreads();
    }
    const float inv = 1.f / red[0];
    for (int j = tid; j < S_; j += 256)
        orow[j] = (j <= q) ? (_Float16)(__expf(row[j] - m) * inv) : (_Float16)0.f;
}

// ---------------------------------------------------------------------------
extern "C" void kernel_launch(void* const* d_in, const int* in_sizes, int n_in,
                              void* d_out, int out_size, void* d_ws, size_t ws_size,
                              hipStream_t stream)
{
    (void)in_sizes; (void)n_in; (void)out_size; (void)ws_size;

    const int*   x      = (const int*)  d_in[0];
    const float* emb    = (const float*)d_in[1];
    const float* Wq     = (const float*)d_in[2];
    const float* Wk     = (const float*)d_in[3];
    const float* Wv     = (const float*)d_in[4];
    const float* Wo     = (const float*)d_in[5];
    const float* bo     = (const float*)d_in[6];
    const float* ln1w   = (const float*)d_in[7];
    const float* ln1b   = (const float*)d_in[8];
    const float* W1     = (const float*)d_in[9];
    const float* b1     = (const float*)d_in[10];
    const float* W2     = (const float*)d_in[11];
    const float* b2     = (const float*)d_in[12];
    const float* ln2w   = (const float*)d_in[13];
    const float* ln2b   = (const float*)d_in[14];
    const float* fnw    = (const float*)d_in[15];
    const float* fnb    = (const float*)d_in[16];
    const float* lmhead = (const float*)d_in[17];
    float* out = (float*)d_out;

    // ---- workspace carve-up (~292 MB total) ----
    char* wsb = (char*)d_ws;
    size_t off = 0;
    auto alloc = [&](size_t bytes) -> void* {
        off = (off + 255) & ~(size_t)255;
        void* p = wsb + off;
        off += bytes;
        return p;
    };
    float*     h      = (float*)    alloc((size_t)S_ * D_ * 4);
    _Float16*  hn16   = (_Float16*) alloc((size_t)S_ * D_ * 2);
    _Float16*  q16    = (_Float16*) alloc((size_t)S_ * D_ * 2);
    _Float16*  k16    = (_Float16*) alloc((size_t)S_ * D_ * 2);
    _Float16*  v16    = (_Float16*) alloc((size_t)S_ * D_ * 2);
    _Float16*  vt16   = (_Float16*) alloc((size_t)S_ * D_ * 2);   // [H][HD][S]
    _Float16*  ao16   = (_Float16*) alloc((size_t)S_ * D_ * 2);
    _Float16*  a116   = (_Float16*) alloc((size_t)S_ * FF_ * 2);
    float*     scores = (float*)    alloc((size_t)H_ * S_ * S_ * 4);
    _Float16*  attn16 = (_Float16*) alloc((size_t)H_ * S_ * S_ * 2);
    _Float16*  wqT    = (_Float16*) alloc((size_t)L_ * D_ * D_ * 2);
    _Float16*  wkT    = (_Float16*) alloc((size_t)L_ * D_ * D_ * 2);
    _Float16*  wvT    = (_Float16*) alloc((size_t)L_ * D_ * D_ * 2);
    _Float16*  woT    = (_Float16*) alloc((size_t)L_ * D_ * D_ * 2);
    _Float16*  w1T    = (_Float16*) alloc((size_t)L_ * D_ * FF_ * 2);
    _Float16*  w2T    = (_Float16*) alloc((size_t)L_ * FF_ * D_ * 2);
    _Float16*  lmT    = (_Float16*) alloc((size_t)D_ * V_ * 2);

    const dim3 tb(32, 8);     // transpose block
    const dim3 gb(256);       // gemm / 1D block

    // ---- pack all weights to transposed f16 (N x K, so B-frags load as b128) ----
    transpose_pack_f32<<<dim3(D_/32,  D_/32,  L_), tb, 0, stream>>>(Wq, wqT, D_, D_,  (long)D_*D_,  (long)D_*D_);
    transpose_pack_f32<<<dim3(D_/32,  D_/32,  L_), tb, 0, stream>>>(Wk, wkT, D_, D_,  (long)D_*D_,  (long)D_*D_);
    transpose_pack_f32<<<dim3(D_/32,  D_/32,  L_), tb, 0, stream>>>(Wv, wvT, D_, D_,  (long)D_*D_,  (long)D_*D_);
    transpose_pack_f32<<<dim3(D_/32,  D_/32,  L_), tb, 0, stream>>>(Wo, woT, D_, D_,  (long)D_*D_,  (long)D_*D_);
    transpose_pack_f32<<<dim3(FF_/32, D_/32,  L_), tb, 0, stream>>>(W1, w1T, D_, FF_, (long)D_*FF_, (long)D_*FF_);
    transpose_pack_f32<<<dim3(D_/32,  FF_/32, L_), tb, 0, stream>>>(W2, w2T, FF_, D_, (long)FF_*D_, (long)FF_*D_);
    transpose_pack_f32<<<dim3(V_/32,  D_/32,  1),  tb, 0, stream>>>(lmhead, lmT, D_, V_, 0, 0);

    // ---- embedding + positional encoding ----
    embed_kernel<<<dim3((S_ * D_) / 256), gb, 0, stream>>>(x, emb, h);

    const float qk_scale = 0.125f;   // 1/sqrt(HD)

    for (int l = 0; l < L_; ++l) {
        const _Float16* wq = wqT + (long)l * D_ * D_;
        const _Float16* wk = wkT + (long)l * D_ * D_;
        const _Float16* wv = wvT + (long)l * D_ * D_;
        const _Float16* wo = woT + (long)l * D_ * D_;
        const _Float16* w1 = w1T + (long)l * D_ * FF_;
        const _Float16* w2 = w2T + (long)l * FF_ * D_;

        // pre-LN
        layernorm_f16<<<dim3(S_), gb, 0, stream>>>(h, ln1w + l * D_, ln1b + l * D_, hn16, D_);

        // Q, K, V projections (no bias in reference)
        gemm_wmma_f16<<<dim3(D_/64, S_/64, 1), gb, 0, stream>>>(
            hn16, wq, nullptr, nullptr, nullptr, q16,
            S_, D_, D_, D_, D_, D_, 0, 0, 0, 1.f, 0);
        gemm_wmma_f16<<<dim3(D_/64, S_/64, 1), gb, 0, stream>>>(
            hn16, wk, nullptr, nullptr, nullptr, k16,
            S_, D_, D_, D_, D_, D_, 0, 0, 0, 1.f, 0);
        gemm_wmma_f16<<<dim3(D_/64, S_/64, 1), gb, 0, stream>>>(
            hn16, wv, nullptr, nullptr, nullptr, v16,
            S_, D_, D_, D_, D_, D_, 0, 0, 0, 1.f, 0);

        // V transpose per head: vt[h][hd][s] = v[s][h*HD+hd]
        transpose_f16<<<dim3(HD_/32, S_/32, H_), tb, 0, stream>>>(
            v16, vt16, D_, S_, (long)HD_, (long)HD_ * S_);

        // scores[h] = Q_h @ K_h^T / sqrt(HD)   (K row-major IS the packed-B layout)
        gemm_wmma_f16<<<dim3(S_/64, S_/64, H_), gb, 0, stream>>>(
            q16, k16, nullptr, nullptr, scores, nullptr,
            S_, S_, HD_, D_, D_, S_, (long)HD_, (long)HD_, (long)S_ * S_, qk_scale, 0);

        // causal softmax -> f16 attention weights (masked cols zeroed)
        softmax_causal<<<dim3(S_, H_), gb, 0, stream>>>(scores, attn16);

        // out_h = attn_h @ V_h   -> ao16 columns [h*HD, h*HD+HD)
        gemm_wmma_f16<<<dim3(HD_/64, S_/64, H_), gb, 0, stream>>>(
            attn16, vt16, nullptr, nullptr, nullptr, ao16,
            S_, HD_, S_, S_, S_, D_, (long)S_ * S_, (long)HD_ * S_, (long)HD_, 1.f, 0);

        // h = ao @ Wo + bo + residual
        gemm_wmma_f16<<<dim3(D_/64, S_/64, 1), gb, 0, stream>>>(
            ao16, wo, bo + l * D_, h, h, nullptr,
            S_, D_, D_, D_, D_, D_, 0, 0, 0, 1.f, 0);

        // FFN
        layernorm_f16<<<dim3(S_), gb, 0, stream>>>(h, ln2w + l * D_, ln2b + l * D_, hn16, D_);
        gemm_wmma_f16<<<dim3(FF_/64, S_/64, 1), gb, 0, stream>>>(
            hn16, w1, b1 + l * FF_, nullptr, nullptr, a116,
            S_, FF_, D_, D_, D_, FF_, 0, 0, 0, 1.f, 1 /*relu*/);
        gemm_wmma_f16<<<dim3(D_/64, S_/64, 1), gb, 0, stream>>>(
            a116, w2, b2 + l * D_, h, h, nullptr,
            S_, D_, FF_, FF_, FF_, D_, 0, 0, 0, 1.f, 0);
    }

    // final LN + lm_head -> d_out [S, V] f32
    layernorm_f16<<<dim3(S_), gb, 0, stream>>>(h, fnw, fnb, hn16, D_);
    gemm_wmma_f16<<<dim3(V_/64, S_/64, 1), gb, 0, stream>>>(
        hn16, lmT, nullptr, nullptr, out, nullptr,
        S_, V_, D_, D_, D_, V_, 0, 0, 0, 1.f, 0);
}
